// GATSurv_7035156431029
// MI455X (gfx1250) — matrix operand order repaired and checked by reading, not compiled, not run on hardware
//
#include <hip/hip_runtime.h>
#include <hip/hip_bf16.h>
#include <stdint.h>

typedef _Float16 v16h __attribute__((ext_vector_type(16)));
typedef _Float16 h8   __attribute__((ext_vector_type(8)));
typedef float    v8f  __attribute__((ext_vector_type(8)));

#define BS  128
#define NN  320
#define NH1 8
#define F1  18
#define D1  144
#define NH2 4
#define F2  46
#define D2  184
#define K2P 160   // D1 padded to multiple of 32 for WMMA k-loop
#define F1P 32    // F1 padded to 2 N-tiles
#define F2P 48    // F2 padded to 3 N-tiles
#define NEGV (-9.0e15f)
#define GATT 320  // threads per GAT block (10 waves: 20 M-tiles / 10 = 2 each)

static __device__ __forceinline__ v8f wmma_f16(v16h a, v16h b, v8f c) {
  return __builtin_amdgcn_wmma_f32_16x16x32_f16(false, a, false, b, (short)0, c,
                                                false, false);
}
static __device__ __forceinline__ float eluf(float x) {
  return x > 0.f ? x : expm1f(x);
}

// ---------------------------------------------------------------------------
// Pack adjacency (adj > 0) into bitmask: adjbits[row*10 + word]
// ---------------------------------------------------------------------------
__global__ __launch_bounds__(256) void k_adjbits(const float* __restrict__ adj,
                                                 uint32_t* __restrict__ adjbits) {
  int idx = blockIdx.x * 256 + threadIdx.x;
  if (idx >= NN * 10) return;
  int i = idx / 10, w = idx % 10;
  uint32_t bits = 0;
  for (int t = 0; t < 32; ++t)
    if (adj[i * NN + w * 32 + t] > 0.f) bits |= (1u << t);
  adjbits[idx] = bits;
}

// ---------------------------------------------------------------------------
// Pre-convert W2 -> f16, transposed [hd][f(48)][k(160)], zero padded
// ---------------------------------------------------------------------------
__global__ __launch_bounds__(256) void k_w2t(const float* __restrict__ W2,
                                             _Float16* __restrict__ w2t) {
  int idx = blockIdx.x * 256 + threadIdx.x;
  if (idx >= NH2 * F2P * K2P) return;
  int hd = idx / (F2P * K2P), rem = idx % (F2P * K2P);
  int f = rem / K2P, k = rem % K2P;
  float v = (f < F2 && k < D1) ? W2[(hd * D1 + k) * F2 + f] : 0.f;
  w2t[idx] = (_Float16)v;
}

// ---------------------------------------------------------------------------
// Layer-1 GAT: one block per (b, head). FIN==1 so h[n][f] = x[b,n]*W1[hd,f]
// and es/ed are scalars times x. Attention aggregation via v_wmma f16->f32.
// ---------------------------------------------------------------------------
__global__ __launch_bounds__(GATT) void k_gat1(
    const float* __restrict__ x, const float* __restrict__ W1,
    const float* __restrict__ a1, const uint32_t* __restrict__ adjbits,
    float* __restrict__ out1) {
  __shared__ _Float16 hB[F1P * NN];       // [f][k] f16, zero padded f>=18
  __shared__ float es_s[NN], ed_s[NN];
  __shared__ uint32_t adjb[NN * 10];

  const int b = blockIdx.x >> 3, hd = blockIdx.x & 7;
  const int tid = threadIdx.x;

  float Ss = 0.f, Sd = 0.f;
#pragma unroll
  for (int f = 0; f < F1; ++f) {
    float w = W1[hd * F1 + f];
    Ss += w * a1[hd * 2 * F1 + f];
    Sd += w * a1[hd * 2 * F1 + F1 + f];
  }
  for (int i = tid; i < NN * 10; i += GATT) adjb[i] = adjbits[i];
  for (int n = tid; n < NN; n += GATT) {
    float xv = x[b * NN + n];
#pragma unroll
    for (int f = 0; f < F1; ++f) hB[f * NN + n] = (_Float16)(xv * W1[hd * F1 + f]);
#pragma unroll
    for (int f = F1; f < F1P; ++f) hB[f * NN + n] = (_Float16)0.f;
    es_s[n] = xv * Ss;
    ed_s[n] = xv * Sd;
  }
  __syncthreads();

  const int wave = tid >> 5, lane = tid & 31, half = lane >> 4, lr = lane & 15;
  for (int mt = wave; mt < NN / 16; mt += GATT / 32) {
    const int i0 = mt * 16, r = i0 + lr;
    const float es_r = es_s[r];
    const uint32_t* arow = &adjb[r * 10];

    // pass 1: row max over this lane's owned half of the columns
    float mx = -3.0e38f;
    for (int jj = 0; jj < NN; jj += 16) {
#pragma unroll
      for (int c = 0; c < 8; ++c) {
        int j = jj + half * 8 + c;
        float e = es_r + ed_s[j];
        e = fmaxf(e, 0.2f * e);
        if (!((arow[j >> 5] >> (j & 31)) & 1u)) e = NEGV;
        mx = fmaxf(mx, e);
      }
    }
    mx = fmaxf(mx, __shfl_xor(mx, 16, 32));
    // pass 2: row sum of exp
    float sum = 0.f;
    for (int jj = 0; jj < NN; jj += 16) {
#pragma unroll
      for (int c = 0; c < 8; ++c) {
        int j = jj + half * 8 + c;
        float e = es_r + ed_s[j];
        e = fmaxf(e, 0.2f * e);
        if (!((arow[j >> 5] >> (j & 31)) & 1u)) e = NEGV;
        sum += __expf(e - mx);
      }
    }
    sum += __shfl_xor(sum, 16, 32);
    const float rinv = 1.f / sum;

    v8f c0 = {}, c1 = {};
    for (int kb = 0; kb < NN; kb += 32) {
      v16h A;
#pragma unroll
      for (int i = 0; i < 16; ++i) {
        // A-fragment K index: contiguous, lane-half gets 16 consecutive K
        int j = kb + half * 16 + i;
        float e = es_r + ed_s[j];
        e = fmaxf(e, 0.2f * e);
        float p = ((arow[j >> 5] >> (j & 31)) & 1u) ? __expf(e - mx) * rinv : 0.f;
        A[i] = (_Float16)p;
      }
      // B-fragment: lane-half gets 16 contiguous K values (32B LDS load)
      v16h B0 = *(const v16h*)&hB[lr * NN + kb + half * 16];
      c0 = wmma_f16(A, B0, c0);
      v16h B1 = *(const v16h*)&hB[(16 + lr) * NN + kb + half * 16];
      c1 = wmma_f16(A, B1, c1);
    }
#pragma unroll
    for (int v = 0; v < 8; ++v) {
      int row = i0 + v + half * 8;
      float* orow = &out1[((size_t)b * NN + row) * D1 + hd * F1];
      orow[lr] = eluf(c0[v]);
      if (lr < F1 - 16) orow[16 + lr] = eluf(c1[v]);
    }
  }
}

// ---------------------------------------------------------------------------
// x0 copy, pool1 (x1), and f32->f16 padded conversion of layer-1 output
// ---------------------------------------------------------------------------
__global__ __launch_bounds__(320) void k_pool1(
    const float* __restrict__ x, const float* __restrict__ out1,
    const float* __restrict__ p1w, const float* __restrict__ p1b,
    float* __restrict__ out, _Float16* __restrict__ out1h) {
  int b = blockIdx.x, n = threadIdx.x;
  const float* row = &out1[((size_t)b * NN + n) * D1];
  _Float16* hrow = &out1h[((size_t)b * NN + n) * K2P];
  out[(size_t)b * 3 * NN + n] = x[b * NN + n];  // x0 = mean over FIN==1
  float s = p1b[0];
  for (int k = 0; k < D1; ++k) {
    float v = row[k];
    s += v * p1w[k];
    hrow[k] = (_Float16)v;
  }
  for (int k = D1; k < K2P; ++k) hrow[k] = (_Float16)0.f;
  out[(size_t)b * 3 * NN + NN + n] = s;
}

// ---------------------------------------------------------------------------
// Layer-2 GAT: one block per (b, head).
// Phase C: h2 = X1 @ W2[hd] via WMMA (A from global f16, B pre-converted).
// Phase D: attention aggregation via WMMA (3 N-tiles).
// es/ed via algebraic shortcut: es = X1 . (W2 @ a_src).
// ---------------------------------------------------------------------------
__global__ __launch_bounds__(GATT) void k_gat2(
    const float* __restrict__ out1, const _Float16* __restrict__ out1h,
    const _Float16* __restrict__ w2t, const float* __restrict__ W2,
    const float* __restrict__ a2, const uint32_t* __restrict__ adjbits,
    float* __restrict__ out2) {
  __shared__ _Float16 hB[F2P * NN];  // h2 f16, [f][k]
  __shared__ float es_s[NN], ed_s[NN];
  __shared__ float wsrc[D1], wdst[D1];
  __shared__ uint32_t adjb[NN * 10];

  const int b = blockIdx.x >> 2, hd = blockIdx.x & 3;
  const int tid = threadIdx.x;

  for (int k = tid; k < D1; k += GATT) {
    float s = 0.f, d = 0.f;
    for (int f = 0; f < F2; ++f) {
      float w = W2[(hd * D1 + k) * F2 + f];
      s += w * a2[hd * 2 * F2 + f];
      d += w * a2[hd * 2 * F2 + F2 + f];
    }
    wsrc[k] = s;
    wdst[k] = d;
  }
  for (int i = tid; i < NN * 10; i += GATT) adjb[i] = adjbits[i];
  __syncthreads();

  for (int n = tid; n < NN; n += GATT) {
    const float* xr = &out1[((size_t)b * NN + n) * D1];
    float s = 0.f, d = 0.f;
    for (int k = 0; k < D1; ++k) {
      float v = xr[k];
      s += v * wsrc[k];
      d += v * wdst[k];
    }
    es_s[n] = s;
    ed_s[n] = d;
  }

  const int wave = tid >> 5, lane = tid & 31, half = lane >> 4, lr = lane & 15;

  // ---- Phase C: h2 tiles (20 m-tiles x 3 n-tiles, K padded to 160) ----
  for (int tile = wave; tile < 60; tile += GATT / 32) {
    int mt = tile / 3, t = tile % 3, i0 = mt * 16, col = t * 16 + lr, r = i0 + lr;
    v8f c = {};
    const _Float16* arow = &out1h[((size_t)b * NN + r) * K2P];
    const _Float16* brow = &w2t[((size_t)hd * F2P + col) * K2P];
    for (int ks = 0; ks < 5; ++ks) {
      int kb = ks * 32;
      h8 a0 = *(const h8*)&arow[kb + half * 16];
      h8 a1v = *(const h8*)&arow[kb + half * 16 + 8];
      v16h A = __builtin_shufflevector(a0, a1v, 0, 1, 2, 3, 4, 5, 6, 7, 8, 9, 10,
                                       11, 12, 13, 14, 15);
      v16h B = *(const v16h*)&brow[kb + half * 16];
      c = wmma_f16(A, B, c);
    }
    h8 st;
#pragma unroll
    for (int v = 0; v < 8; ++v) st[v] = (_Float16)c[v];
    *(h8*)&hB[col * NN + i0 + half * 8] = st;  // transposed store [f][k]
  }
  __syncthreads();

  // ---- Phase D: attention aggregation ----
  for (int mt = wave; mt < NN / 16; mt += GATT / 32) {
    const int i0 = mt * 16, r = i0 + lr;
    const float es_r = es_s[r];
    const uint32_t* arow = &adjb[r * 10];

    float mx = -3.0e38f;
    for (int jj = 0; jj < NN; jj += 16) {
#pragma unroll
      for (int c = 0; c < 8; ++c) {
        int j = jj + half * 8 + c;
        float e = es_r + ed_s[j];
        e = fmaxf(e, 0.2f * e);
        if (!((arow[j >> 5] >> (j & 31)) & 1u)) e = NEGV;
        mx = fmaxf(mx, e);
      }
    }
    mx = fmaxf(mx, __shfl_xor(mx, 16, 32));
    float sum = 0.f;
    for (int jj = 0; jj < NN; jj += 16) {
#pragma unroll
      for (int c = 0; c < 8; ++c) {
        int j = jj + half * 8 + c;
        float e = es_r + ed_s[j];
        e = fmaxf(e, 0.2f * e);
        if (!((arow[j >> 5] >> (j & 31)) & 1u)) e = NEGV;
        sum += __expf(e - mx);
      }
    }
    sum += __shfl_xor(sum, 16, 32);
    const float rinv = 1.f / sum;

    v8f c0 = {}, c1 = {}, c2 = {};
    for (int kb = 0; kb < NN; kb += 32) {
      v16h A;
#pragma unroll
      for (int i = 0; i < 16; ++i) {
        int j = kb + half * 16 + i;
        float e = es_r + ed_s[j];
        e = fmaxf(e, 0.2f * e);
        float p = ((arow[j >> 5] >> (j & 31)) & 1u) ? __expf(e - mx) * rinv : 0.f;
        A[i] = (_Float16)p;
      }
      v16h B0 = *(const v16h*)&hB[lr * NN + kb + half * 16];
      c0 = wmma_f16(A, B0, c0);
      v16h B1 = *(const v16h*)&hB[(16 + lr) * NN + kb + half * 16];
      c1 = wmma_f16(A, B1, c1);
      v16h B2 = *(const v16h*)&hB[(32 + lr) * NN + kb + half * 16];
      c2 = wmma_f16(A, B2, c2);
    }
#pragma unroll
    for (int v = 0; v < 8; ++v) {
      int row = i0 + v + half * 8;
      float* orow = &out2[((size_t)b * NN + row) * D2 + hd * F2];
      orow[lr] = eluf(c0[v]);
      orow[16 + lr] = eluf(c1[v]);
      if (lr < F2 - 32) orow[32 + lr] = eluf(c2[v]);
    }
  }
}

// ---------------------------------------------------------------------------
// pool2 -> x2 section of x_concat
// ---------------------------------------------------------------------------
__global__ __launch_bounds__(320) void k_pool2(const float* __restrict__ out2,
                                               const float* __restrict__ p2w,
                                               const float* __restrict__ p2b,
                                               float* __restrict__ out) {
  int b = blockIdx.x, n = threadIdx.x;
  const float* row = &out2[((size_t)b * NN + n) * D2];
  float s = p2b[0];
  for (int k = 0; k < D2; ++k) s += row[k] * p2w[k];
  out[(size_t)b * 3 * NN + 2 * NN + n] = s;
}

// ---------------------------------------------------------------------------
// MLP head: 960 -> 104 -> 45 -> 84 (elu each) -> hazard scalar
// ---------------------------------------------------------------------------
__global__ __launch_bounds__(128) void k_mlp(
    const float* __restrict__ xc_all, const float* __restrict__ e1w,
    const float* __restrict__ e1b, const float* __restrict__ e2w,
    const float* __restrict__ e2b, const float* __restrict__ e3w,
    const float* __restrict__ e3b, const float* __restrict__ cw,
    const float* __restrict__ cb, float* __restrict__ feat_out,
    float* __restrict__ haz_out) {
  __shared__ float xc[3 * NN];
  __shared__ float h1s[104];
  __shared__ float h2s[45];
  __shared__ float f3[84];
  int b = blockIdx.x, t = threadIdx.x;
  for (int i = t; i < 3 * NN; i += 128) xc[i] = xc_all[(size_t)b * 3 * NN + i];
  __syncthreads();
  for (int j = t; j < 104; j += 128) {
    float s = e1b[j];
    for (int i = 0; i < 3 * NN; ++i) s += xc[i] * e1w[i * 104 + j];
    h1s[j] = eluf(s);
  }
  __syncthreads();
  for (int j = t; j < 45; j += 128) {
    float s = e2b[j];
    for (int i = 0; i < 104; ++i) s += h1s[i] * e2w[i * 45 + j];
    h2s[j] = eluf(s);
  }
  __syncthreads();
  for (int j = t; j < 84; j += 128) {
    float s = e3b[j];
    for (int i = 0; i < 45; ++i) s += h2s[i] * e3w[i * 84 + j];
    float v = eluf(s);
    f3[j] = v;
    feat_out[(size_t)b * 84 + j] = v;
  }
  __syncthreads();
  if (t == 0) {
    float s = cb[0];
    for (int i = 0; i < 84; ++i) s += f3[i] * cw[i];
    haz_out[b] = s;
  }
}

// ---------------------------------------------------------------------------
extern "C" void kernel_launch(void* const* d_in, const int* in_sizes, int n_in,
                              void* d_out, int out_size, void* d_ws,
                              size_t ws_size, hipStream_t stream) {
  (void)in_sizes; (void)n_in; (void)out_size; (void)ws_size;
  const float* x   = (const float*)d_in[0];
  const float* adj = (const float*)d_in[1];
  const float* W1  = (const float*)d_in[2];
  const float* a1  = (const float*)d_in[3];
  const float* W2  = (const float*)d_in[4];
  const float* a2  = (const float*)d_in[5];
  const float* p1w = (const float*)d_in[6];
  const float* p1b = (const float*)d_in[7];
  const float* p2w = (const float*)d_in[8];
  const float* p2b = (const float*)d_in[9];
  const float* e1w = (const float*)d_in[10];
  const float* e1b = (const float*)d_in[11];
  const float* e2w = (const float*)d_in[12];
  const float* e2b = (const float*)d_in[13];
  const float* e3w = (const float*)d_in[14];
  const float* e3b = (const float*)d_in[15];
  const float* cw  = (const float*)d_in[16];
  const float* cb  = (const float*)d_in[17];
  float* out = (float*)d_out;

  char* ws = (char*)d_ws;
  size_t o = 0;
  uint32_t* adjbits = (uint32_t*)(ws + o); o += (size_t)NN * 10 * 4;         // 12800
  _Float16* w2t     = (_Float16*)(ws + o); o += (size_t)NH2 * F2P * K2P * 2; // 61440
  float*    out1    = (float*)(ws + o);    o += (size_t)BS * NN * D1 * 4;    // 23.6MB
  _Float16* out1h   = (_Float16*)(ws + o); o += (size_t)BS * NN * K2P * 2;   // 13.1MB
  float*    out2    = (float*)(ws + o);    o += (size_t)BS * NN * D2 * 4;    // 30.1MB

  k_adjbits<<<(NN * 10 + 255) / 256, 256, 0, stream>>>(adj, adjbits);
  k_w2t<<<(NH2 * F2P * K2P + 255) / 256, 256, 0, stream>>>(W2, w2t);
  k_gat1<<<BS * NH1, GATT, 0, stream>>>(x, W1, a1, adjbits, out1);
  k_pool1<<<BS, NN, 0, stream>>>(x, out1, p1w, p1b, out, out1h);
  k_gat2<<<BS * NH2, GATT, 0, stream>>>(out1, out1h, w2t, W2, a2, adjbits, out2);
  k_pool2<<<BS, NN, 0, stream>>>(out2, p2w, p2b, out);
  k_mlp<<<BS, 128, 0, stream>>>(out, e1w, e1b, e2w, e2b, e3w, e3b, cw, cb,
                                out + (size_t)BS * 3 * NN,
                                out + (size_t)BS * 3 * NN + (size_t)BS * 84);
}
